// TransformerBlock_25168508354973
// MI455X (gfx1250) — matrix operand order repaired and checked by reading, not compiled, not run on hardware
//
#include <hip/hip_runtime.h>
#include <hip/hip_bf16.h>
#include <math.h>

// ---------------------------------------------------------------------------
// Transformer block for MI455X (gfx1250, wave32, WMMA + async-LDS staging).
// ---------------------------------------------------------------------------

typedef __attribute__((ext_vector_type(16))) __bf16 v16bf;
typedef __attribute__((ext_vector_type(8)))  float  v8f;

#define EMBED   1024
#define HEADS   16
#define HEAD_DIM 64
#define HIDDEN  4096
#define BATCH   2
#define SEQ     2048
#define NROWS   (BATCH * SEQ)          // 4096 token rows

__device__ __forceinline__ v8f wmma_bf16(v16bf a, v16bf b, v8f c) {
  // D = A(16x32 bf16) * B(32x16 bf16) + C(16x16 f32)
  return __builtin_amdgcn_wmma_f32_16x16x32_bf16(
      /*neg_a=*/false, a, /*neg_b=*/false, b,
      /*c_mod=*/(short)0, c, /*reuse_a=*/false, /*reuse_b=*/false);
}

// A-fragment loader: 16x32 bf16 tile at (row0, k0) from row-major [*, ld].
// ISA layout: lane l holds row (l&15); VGPR v holds K = 2v + (v>=4?8:0) + (l>=16?8:0).
__device__ __forceinline__ v16bf load_a_frag(const __bf16* base, int ld,
                                             int row0, int k0, int lane) {
  union { v16bf v; unsigned int u[8]; } f;
  const int m   = row0 + (lane & 15);
  const int khi = (lane >= 16) ? 8 : 0;
#pragma unroll
  for (int v = 0; v < 8; ++v) {
    const int k = k0 + 2 * v + ((v >= 4) ? 8 : 0) + khi;
    f.u[v] = *reinterpret_cast<const unsigned int*>(base + (size_t)m * ld + k);
  }
  return f.v;
}

// B-fragment loader: 32x16 bf16 tile, operand stored K-contiguous as [col][K].
// ISA layout: lane l holds col (l&15); VGPR v holds K = 2v + (v>=4?8:0) + (l>=16?8:0).
__device__ __forceinline__ v16bf load_b_frag(const __bf16* base, int ld,
                                             int col0, int k0, int lane) {
  union { v16bf v; unsigned int u[8]; } f;
  const int n   = col0 + (lane & 15);
  const int khi = (lane >= 16) ? 8 : 0;
#pragma unroll
  for (int v = 0; v < 8; ++v) {
    const int k = k0 + 2 * v + ((v >= 4) ? 8 : 0) + khi;
    f.u[v] = *reinterpret_cast<const unsigned int*>(base + (size_t)n * ld + k);
  }
  return f.v;
}

// Async DMA of one 32x32 bf16 A-tile (2 KB) into LDS, spread over the
// 128 threads of the block: each lane moves 16 bytes.  ASYNCcnt-tracked.
__device__ __forceinline__ void async_copy_a_tile(const __bf16* A, int K, int tm,
                                                  int kk, __bf16* lds, int tid) {
  const int r  = tid >> 2;        // 0..31  (tile row)
  const int c4 = tid & 3;         // 0..3   (16B chunk within row)
  const unsigned long long ga =
      (unsigned long long)(const void*)(A + (size_t)(tm + r) * K + kk + c4 * 8);
  const unsigned ldsoff = (unsigned)(unsigned long long)(lds + r * 32 + c4 * 8);
  asm volatile("global_load_async_to_lds_b128 %0, %1, off"
               :: "v"(ldsoff), "v"(ga) : "memory");
}

__device__ __forceinline__ void wait_async_le1() {
  asm volatile("s_wait_asynccnt 0x1" ::: "memory");
}
__device__ __forceinline__ void wait_async_le0() {
  asm volatile("s_wait_asynccnt 0x0" ::: "memory");
}

// ---------------------------------------------------------------------------
// Weight convert+transpose: w[K][N] f32 -> wT[N][K] bf16
// ---------------------------------------------------------------------------
__global__ void transpose_to_bf16(const float* __restrict__ w,
                                  __bf16* __restrict__ wT, int K, int N) {
  const size_t idx = (size_t)blockIdx.x * blockDim.x + threadIdx.x;
  const size_t total = (size_t)K * N;
  if (idx >= total) return;
  const int n = (int)(idx / K);
  const int k = (int)(idx % K);
  wT[idx] = (__bf16)w[(size_t)k * N + n];
}

// ---------------------------------------------------------------------------
// LayerNorm: f32 row -> bf16 normalized row.  One 256-thread block per row.
// ---------------------------------------------------------------------------
__global__ void layernorm_bf16(const float* __restrict__ x,
                               const float* __restrict__ g,
                               const float* __restrict__ b,
                               __bf16* __restrict__ out, int C) {
  __shared__ float r1[256];
  __shared__ float r2[256];
  const int row = blockIdx.x;
  const int tid = threadIdx.x;
  float s = 0.f, ss = 0.f;
  for (int i = tid; i < C; i += 256) {
    const float v = x[(size_t)row * C + i];
    s += v; ss += v * v;
  }
  r1[tid] = s; r2[tid] = ss;
  __syncthreads();
  for (int off = 128; off > 0; off >>= 1) {
    if (tid < off) { r1[tid] += r1[tid + off]; r2[tid] += r2[tid + off]; }
    __syncthreads();
  }
  const float mean = r1[0] / C;
  const float var  = r2[0] / C - mean * mean;
  const float rstd = rsqrtf(var + 1e-5f);
  for (int i = tid; i < C; i += 256) {
    const float v = (x[(size_t)row * C + i] - mean) * rstd * g[i] + b[i];
    out[(size_t)row * C + i] = (__bf16)v;
  }
}

// ---------------------------------------------------------------------------
// WMMA GEMM with async A-tile staging.
// Block = 128 threads = 4 waves; macro-tile 32(M) x 128(N); each wave does
// a 32x32 output tile.  The shared 32x32 A tile is DMAed into LDS with
// global_load_async_to_lds_b128, double buffered, reused by all 4 waves.
//   mode 0: QKV scatter  -> q[BH,T,64], k[BH,T,64], vT[BH,64,T] (bf16)
//   mode 1: f32 out      -> out0[row,N] = resid[row,N] + val
//   mode 2: bf16 out     -> out0[row,N] = gelu_exact(val)
// ---------------------------------------------------------------------------
__global__ void gemm_wmma(const __bf16* __restrict__ A,
                          const __bf16* __restrict__ WT,
                          const float* __restrict__ bias,
                          const float* __restrict__ resid,
                          void* __restrict__ out0,
                          int M, int N, int K, int Tdim, int mode) {
  __shared__ __attribute__((aligned(16))) __bf16 lds_a[2][32 * 32];

  const int tid  = threadIdx.x;        // 0..127
  const int lane = tid & 31;
  const int w    = tid >> 5;           // wave 0..3
  const int tilesN = N >> 7;           // 128-wide macro tiles
  const int bm = blockIdx.x / tilesN;
  const int bn = blockIdx.x % tilesN;
  const int tm = bm << 5;
  const int tn = (bn << 7) + (w << 5);

  v8f acc[2][2];
#pragma unroll
  for (int i = 0; i < 2; ++i)
#pragma unroll
    for (int j = 0; j < 2; ++j)
#pragma unroll
      for (int r = 0; r < 8; ++r) acc[i][j][r] = 0.f;

  // prologue: DMA first A tile
  async_copy_a_tile(A, K, tm, 0, lds_a[0], tid);

  int cur = 0;
  for (int kk = 0; kk < K; kk += 32) {
    const bool more = (kk + 32) < K;
    if (more) async_copy_a_tile(A, K, tm, kk + 32, lds_a[cur ^ 1], tid);
    if (more) wait_async_le1(); else wait_async_le0();
    __syncthreads();                               // tile 'cur' visible block-wide

    // prefetch next-next B stripe into L2
    if (kk + 64 < K)
      __builtin_prefetch(WT + (size_t)(tn + lane) * K + kk + 64, 0, 1);

    const v16bf a0 = load_a_frag(lds_a[cur], 32, 0,  0, lane);
    const v16bf a1 = load_a_frag(lds_a[cur], 32, 16, 0, lane);
    const v16bf b0 = load_b_frag(WT, K, tn,      kk, lane);
    const v16bf b1 = load_b_frag(WT, K, tn + 16, kk, lane);
    acc[0][0] = wmma_bf16(a0, b0, acc[0][0]);
    acc[0][1] = wmma_bf16(a0, b1, acc[0][1]);
    acc[1][0] = wmma_bf16(a1, b0, acc[1][0]);
    acc[1][1] = wmma_bf16(a1, b1, acc[1][1]);

    __syncthreads();                               // done reading before rewrite
    cur ^= 1;
  }

  const int rhi = (lane >= 16) ? 8 : 0;
  const int cl  = lane & 15;
#pragma unroll
  for (int i = 0; i < 2; ++i) {
#pragma unroll
    for (int j = 0; j < 2; ++j) {
#pragma unroll
      for (int r = 0; r < 8; ++r) {
        const int row = tm + i * 16 + r + rhi;
        const int col = tn + j * 16 + cl;
        float val = acc[i][j][r] + bias[col];
        if (mode == 0) {
          // QKV scatter
          const int part = col >> 10;       // 0=q 1=k 2=v
          const int cc   = col & 1023;
          const int h    = cc >> 6;
          const int d    = cc & 63;
          const int bb   = row / Tdim;
          const int t    = row % Tdim;
          const int bh   = bb * HEADS + h;
          __bf16* base = (__bf16*)out0 + (size_t)part * ((size_t)M * EMBED);
          if (part < 2)
            base[((size_t)bh * Tdim + t) * HEAD_DIM + d] = (__bf16)val;
          else
            base[((size_t)bh * HEAD_DIM + d) * Tdim + t] = (__bf16)val;  // V^T
        } else if (mode == 1) {
          ((float*)out0)[(size_t)row * N + col] =
              resid[(size_t)row * N + col] + val;
        } else {
          const float gv = 0.5f * val * (1.0f + erff(val * 0.70710678118654752f));
          ((__bf16*)out0)[(size_t)row * N + col] = (__bf16)gv;
        }
      }
    }
  }
}

// ---------------------------------------------------------------------------
// Flash attention, causal.  One wave per 16-query tile, 32-key blocks,
// online softmax.  q,k: [BH,T,64] bf16; vT: [BH,64,T] bf16; out: [B,T,C] bf16.
// ---------------------------------------------------------------------------
__device__ __forceinline__ float rowmax16(float v) {
#pragma unroll
  for (int off = 1; off < 16; off <<= 1) v = fmaxf(v, __shfl_xor(v, off, 32));
  return v;
}
__device__ __forceinline__ float rowsum16(float v) {
#pragma unroll
  for (int off = 1; off < 16; off <<= 1) v += __shfl_xor(v, off, 32);
  return v;
}

__global__ void attn_flash(const __bf16* __restrict__ q,
                           const __bf16* __restrict__ kmat,
                           const __bf16* __restrict__ vT,
                           __bf16* __restrict__ attn, int T) {
  __shared__ __attribute__((aligned(16))) __bf16 lds_p[16 * 32];
  const int lane = threadIdx.x;            // blockDim.x == 32
  const int tile = blockIdx.x;             // B*H*(T/16)
  const int qt   = tile % (T / 16);
  const int bh   = tile / (T / 16);
  const int bb   = bh / HEADS;
  const int h    = bh % HEADS;
  const int q0   = qt * 16;

  const __bf16* qbase = q    + (size_t)bh * T * HEAD_DIM;
  const __bf16* kbase = kmat + (size_t)bh * T * HEAD_DIM;
  const __bf16* vbase = vT   + (size_t)bh * HEAD_DIM * T;

  const v16bf aq0 = load_a_frag(qbase, HEAD_DIM, q0, 0,  lane);
  const v16bf aq1 = load_a_frag(qbase, HEAD_DIM, q0, 32, lane);

  v8f o[4];
  float mrow[8], lrow[8];
#pragma unroll
  for (int nt = 0; nt < 4; ++nt)
#pragma unroll
    for (int r = 0; r < 8; ++r) o[nt][r] = 0.f;
#pragma unroll
  for (int r = 0; r < 8; ++r) { mrow[r] = -1e30f; lrow[r] = 0.f; }

  const int rhi = (lane >= 16) ? 8 : 0;
  const int cl  = lane & 15;
  const float scale = 0.125f;              // 1/sqrt(64)
  const int jmax = (q0 + 15) / 32;

  for (int j = 0; j <= jmax; ++j) {
    const int kb = j * 32;
    v8f s0, s1;
#pragma unroll
    for (int r = 0; r < 8; ++r) { s0[r] = 0.f; s1[r] = 0.f; }
    // S = Q * K^T for keys [kb, kb+32)
    s0 = wmma_bf16(aq0, load_b_frag(kbase, HEAD_DIM, kb,      0,  lane), s0);
    s0 = wmma_bf16(aq1, load_b_frag(kbase, HEAD_DIM, kb,      32, lane), s0);
    s1 = wmma_bf16(aq0, load_b_frag(kbase, HEAD_DIM, kb + 16, 0,  lane), s1);
    s1 = wmma_bf16(aq1, load_b_frag(kbase, HEAD_DIM, kb + 16, 32, lane), s1);

    float corr[8];
#pragma unroll
    for (int r = 0; r < 8; ++r) {
      const int row = q0 + r + rhi;
      const int c0  = kb + cl;
      const int c1  = c0 + 16;
      float x0 = s0[r] * scale; if (c0 > row) x0 = -1e30f;
      float x1 = s1[r] * scale; if (c1 > row) x1 = -1e30f;
      const float mx = fmaxf(rowmax16(fmaxf(x0, x1)), mrow[r]);
      const float p0 = __expf(x0 - mx);
      const float p1 = __expf(x1 - mx);
      corr[r] = __expf(mrow[r] - mx);
      lrow[r] = lrow[r] * corr[r] + rowsum16(p0 + p1);
      mrow[r] = mx;
      const int lr = r + rhi;              // local row 0..15
      lds_p[lr * 32 + cl]      = (__bf16)p0;
      lds_p[lr * 32 + 16 + cl] = (__bf16)p1;
    }
    asm volatile("s_wait_dscnt 0x0" ::: "memory");
    const v16bf ap = load_a_frag((const __bf16*)lds_p, 32, 0, 0, lane);
#pragma unroll
    for (int nt = 0; nt < 4; ++nt) {
#pragma unroll
      for (int r = 0; r < 8; ++r) o[nt][r] *= corr[r];
      o[nt] = wmma_bf16(ap, load_b_frag(vbase, T, nt * 16, kb, lane), o[nt]);
    }
  }

  // write out: attn[b, t, h*64 + d]
#pragma unroll
  for (int nt = 0; nt < 4; ++nt) {
#pragma unroll
    for (int r = 0; r < 8; ++r) {
      const int t   = q0 + r + rhi;
      const int col = h * HEAD_DIM + nt * 16 + cl;
      attn[((size_t)bb * T + t) * EMBED + col] = (__bf16)(o[nt][r] / lrow[r]);
    }
  }
}

// ---------------------------------------------------------------------------
// Host launch
// ---------------------------------------------------------------------------
extern "C" void kernel_launch(void* const* d_in, const int* in_sizes, int n_in,
                              void* d_out, int out_size, void* d_ws, size_t ws_size,
                              hipStream_t stream) {
  const float* x      = (const float*)d_in[0];
  const float* ln1_g  = (const float*)d_in[1];
  const float* ln1_b  = (const float*)d_in[2];
  const float* qkv_w  = (const float*)d_in[3];
  const float* qkv_b  = (const float*)d_in[4];
  const float* out_w  = (const float*)d_in[5];
  const float* out_b  = (const float*)d_in[6];
  const float* ln2_g  = (const float*)d_in[7];
  const float* ln2_b  = (const float*)d_in[8];
  const float* fc1_w  = (const float*)d_in[9];
  const float* fc1_b  = (const float*)d_in[10];
  const float* fc2_w  = (const float*)d_in[11];
  const float* fc2_b  = (const float*)d_in[12];
  float* outp = (float*)d_out;

  // workspace carve-up (256B aligned)
  size_t off = 0;
  auto carve = [&](size_t bytes) {
    void* p = (char*)d_ws + off;
    off += (bytes + 255) & ~(size_t)255;
    return p;
  };
  __bf16* wqkvT = (__bf16*)carve((size_t)3072 * 1024 * 2);
  __bf16* woutT = (__bf16*)carve((size_t)1024 * 1024 * 2);
  __bf16* wfc1T = (__bf16*)carve((size_t)4096 * 1024 * 2);
  __bf16* wfc2T = (__bf16*)carve((size_t)1024 * 4096 * 2);
  __bf16* xn1   = (__bf16*)carve((size_t)NROWS * EMBED * 2);
  __bf16* qkvb  = (__bf16*)carve((size_t)3 * NROWS * EMBED * 2);  // q | k | vT
  __bf16* attnb = (__bf16*)carve((size_t)NROWS * EMBED * 2);
  float*  xmid  = (float*) carve((size_t)NROWS * EMBED * 4);
  __bf16* xn2   = (__bf16*)carve((size_t)NROWS * EMBED * 2);
  __bf16* hbuf  = (__bf16*)carve((size_t)NROWS * HIDDEN * 2);

  __bf16* qb  = qkvb;
  __bf16* kb  = qkvb + (size_t)NROWS * EMBED;
  __bf16* vTb = qkvb + (size_t)2 * NROWS * EMBED;

  // 1) weights -> bf16 transposed [N][K]
  transpose_to_bf16<<<(3072 * 1024 + 255) / 256, 256, 0, stream>>>(qkv_w, wqkvT, 1024, 3072);
  transpose_to_bf16<<<(1024 * 1024 + 255) / 256, 256, 0, stream>>>(out_w, woutT, 1024, 1024);
  transpose_to_bf16<<<(4096 * 1024 + 255) / 256, 256, 0, stream>>>(fc1_w, wfc1T, 1024, 4096);
  transpose_to_bf16<<<(4096 * 1024 + 255) / 256, 256, 0, stream>>>(fc2_w, wfc2T, 4096, 1024);

  // 2) LN1
  layernorm_bf16<<<NROWS, 256, 0, stream>>>(x, ln1_g, ln1_b, xn1, EMBED);

  // 3) QKV GEMM (M=4096, N=3072, K=1024): 128 x 24 macro-tiles
  gemm_wmma<<<(NROWS / 32) * (3072 / 128), 128, 0, stream>>>(
      xn1, wqkvT, qkv_b, nullptr, qkvb, NROWS, 3072, 1024, SEQ, 0);

  // 4) flash attention: B*H*(T/16) waves
  attn_flash<<<BATCH * HEADS * (SEQ / 16), 32, 0, stream>>>(qb, kb, vTb, attnb, SEQ);

  // 5) out-proj + residual(x) -> xmid (f32)
  gemm_wmma<<<(NROWS / 32) * (1024 / 128), 128, 0, stream>>>(
      attnb, woutT, out_b, x, xmid, NROWS, 1024, 1024, SEQ, 1);

  // 6) LN2
  layernorm_bf16<<<NROWS, 256, 0, stream>>>(xmid, ln2_g, ln2_b, xn2, EMBED);

  // 7) FC1 + exact GELU -> hbuf (bf16)   (M=4096, N=4096, K=1024)
  gemm_wmma<<<(NROWS / 32) * (HIDDEN / 128), 128, 0, stream>>>(
      xn2, wfc1T, fc1_b, nullptr, hbuf, NROWS, HIDDEN, 1024, SEQ, 2);

  // 8) FC2 + residual(xmid) -> d_out (f32)  (M=4096, N=1024, K=4096)
  gemm_wmma<<<(NROWS / 32) * (1024 / 128), 128, 0, stream>>>(
      hbuf, wfc2T, fc2_b, xmid, outp, NROWS, 1024, 4096, SEQ, 1);
}
